// MultiHeadAttention_21887153340496
// MI455X (gfx1250) — matrix operand order repaired and checked by reading, not compiled, not run on hardware
//
#include <hip/hip_runtime.h>
#include <hip/hip_bf16.h>

// Problem constants (reference: BS=2, S=2048, D=1024, H=16, DK=64)
#define BSZ 2
#define SEQ 2048
#define DIM 1024
#define NH  16
#define DKH 64

typedef __attribute__((ext_vector_type(16))) __bf16 bf16x16;
typedef __attribute__((ext_vector_type(8)))  __bf16 bf16x8;
typedef __attribute__((ext_vector_type(8)))  float  f32x8;

union FragAB { bf16x16 v; bf16x8 h[2]; };

__device__ __forceinline__ f32x8 wmma_bf16(const FragAB& a, const FragAB& b, f32x8 c) {
  // D = A(16x32) * B(32x16) + C, f32 accumulate
  return __builtin_amdgcn_wmma_f32_16x16x32_bf16(false, a.v, false, b.v, (short)0, c,
                                                 false, false);
}

// ---- CDNA5 async global->LDS copy path (probe via __has_builtin) -----------
#ifndef __has_builtin
#define __has_builtin(x) 0
#endif
#if __has_builtin(__builtin_amdgcn_global_load_async_to_lds_b128)
#define HAVE_ASYNC_COPY 1
#else
#define HAVE_ASYNC_COPY 0
#endif

typedef int v4i __attribute__((vector_size(16)));
typedef __attribute__((address_space(1))) v4i* gv4p;  // global int4*
typedef __attribute__((address_space(3))) v4i* lv4p;  // LDS int4*

__device__ __forceinline__ void async_copy16(const void* gsrc, void* ldst) {
#if HAVE_ASYNC_COPY
  __builtin_amdgcn_global_load_async_to_lds_b128((gv4p)(v4i*)gsrc, (lv4p)(v4i*)ldst, 0, 0);
#else
  *(uint4*)ldst = *(const uint4*)gsrc;
#endif
}

__device__ __forceinline__ void async_wait_and_barrier() {
#if HAVE_ASYNC_COPY
#if __has_builtin(__builtin_amdgcn_s_wait_asynccnt)
  __builtin_amdgcn_s_wait_asynccnt(0);
#else
  asm volatile("s_wait_asynccnt 0x0" ::: "memory");
#endif
#endif
  __syncthreads();
}

// ---------------------------------------------------------------------------
// fp32 -> bf16 elementwise convert
// ---------------------------------------------------------------------------
__global__ __launch_bounds__(256)
void cvt_f32_bf16_kernel(const float* __restrict__ in, __bf16* __restrict__ out, int n) {
  int i = blockIdx.x * 256 + threadIdx.x;
  if (i < n) out[i] = (__bf16)in[i];
}

// ---------------------------------------------------------------------------
// fp32 [K=DIM][N=DIM] -> bf16 transposed Wt[n][k]  (coalesced writes)
// ---------------------------------------------------------------------------
__global__ __launch_bounds__(256)
void transpose_bf16_kernel(const float* __restrict__ W, __bf16* __restrict__ Wt) {
  int i = blockIdx.x * 256 + threadIdx.x;   // over DIM*DIM
  int k = i & (DIM - 1);
  int n = i >> 10;
  Wt[(size_t)n * DIM + k] = (__bf16)W[(size_t)k * DIM + n];
}

// ---------------------------------------------------------------------------
// GEMM: Out = A[M x K] * Bt[n][k]^T + bias, M = BSZ*SEQ, N = K = DIM
// mode 0: out bf16, head-split [b][h][s][dk]
// mode 1: out bf16, head-split V transposed [b][h][dk][s]
// mode 2: out f32, plain row-major [m][n]
// block = 128 threads (4 waves); block tile 64x64, wave tile 16x64.
// All fragments for a k-step are loaded before the WMMA chain so the compiler
// can clause the b128 loads and issue the 4 WMMAs back-to-back.
// ---------------------------------------------------------------------------
__global__ __launch_bounds__(128)
void gemm_bf16_kernel(const __bf16* __restrict__ A, const __bf16* __restrict__ Bt,
                      const float* __restrict__ bias, void* __restrict__ Out, int mode) {
  const int K = DIM;
  int tid  = threadIdx.x;
  int wave = tid >> 5;
  int lane = tid & 31;
  int nloc = lane & 15;
  int half = lane >> 4;

  int m0 = blockIdx.x * 64 + wave * 16;
  int n0 = blockIdx.y * 64;

  const __bf16* Arow = A + (size_t)(m0 + nloc) * K;   // A-frag row for this lane
  const __bf16* Brow[4];
  float bia[4];
#pragma unroll
  for (int t = 0; t < 4; ++t) {
    Brow[t] = Bt + (size_t)(n0 + 16 * t + nloc) * K;  // B-frag "column" row (transposed)
    bia[t]  = bias[n0 + 16 * t + nloc];
  }

  f32x8 acc[4] = {};
  for (int k0 = 0; k0 < K; k0 += 32) {
    FragAB a, bf[4];
    // A layout: half-lane K split in 16-chunks
    a.h[0] = *(const bf16x8*)(Arow + k0 + 16 * half);
    a.h[1] = *(const bf16x8*)(Arow + k0 + 16 * half + 8);
    // B layout: half-lane K split in 16-chunks
#pragma unroll
    for (int t = 0; t < 4; ++t) {
      bf[t].h[0] = *(const bf16x8*)(Brow[t] + k0 + 16 * half);
      bf[t].h[1] = *(const bf16x8*)(Brow[t] + k0 + 16 * half + 8);
    }
#pragma unroll
    for (int t = 0; t < 4; ++t) acc[t] = wmma_bf16(a, bf[t], acc[t]);
  }

  // Epilogue: C layout — lane = col (n0+16t+nloc), VGPR j = row (j + 8*half)
#pragma unroll
  for (int t = 0; t < 4; ++t) {
#pragma unroll
    for (int j = 0; j < 8; ++j) {
      float v = acc[t][j] + bia[t];
      int m = m0 + j + 8 * half;
      int n = n0 + 16 * t + nloc;
      if (mode == 2) {
        ((float*)Out)[(size_t)m * DIM + n] = v;
      } else {
        int b  = m >> 11;          // m / SEQ
        int s  = m & (SEQ - 1);
        int hh = n >> 6;           // n / DKH
        int dk = n & (DKH - 1);
        size_t idx = (mode == 0)
                   ? ((((size_t)b * NH + hh) * SEQ + s) * DKH + dk)
                   : ((((size_t)b * NH + hh) * DKH + dk) * SEQ + s);
        ((__bf16*)Out)[idx] = (__bf16)v;
      }
    }
  }
}

// ---------------------------------------------------------------------------
// Flash attention: per (b,h), 64 q-rows per block (16 per wave), 32-key tiles,
// double-buffered K/V staging through LDS via async global->LDS copies.
// Qh, Kh: [b][h][s][dk] bf16 ; Vt: [b][h][dk][s] bf16 ; mask: [b][q][k] int32
// Obf out: [b][s][h*DKH + dk] bf16 (concat layout for final projection)
// ---------------------------------------------------------------------------
#define KS_STRIDE 72   // 32x64 K tile, padded (144B rows, 16B aligned)
#define VS_STRIDE 56   // 64x32 Vt tile, padded (112B rows, 16B aligned)
#define PS_STRIDE 56   // 16x32 per-wave P slab, padded

__global__ __launch_bounds__(128)
void attn_kernel(const __bf16* __restrict__ Qh, const __bf16* __restrict__ Kh,
                 const __bf16* __restrict__ Vt, const int* __restrict__ mask,
                 __bf16* __restrict__ Obf) {
  __shared__ __align__(16) __bf16 Ks[2][32 * KS_STRIDE];
  __shared__ __align__(16) __bf16 Vs[2][64 * VS_STRIDE];
  __shared__ __align__(16) __bf16 Ps[4 * 16 * PS_STRIDE];

  int tid  = threadIdx.x;
  int wave = tid >> 5;
  int lane = tid & 31;
  int nloc = lane & 15;
  int half = lane >> 4;

  int b  = blockIdx.z;
  int hh = blockIdx.y;
  int q0 = blockIdx.x * 64;

  const __bf16* Qbase = Qh + (((size_t)b * NH + hh) * SEQ) * DKH;
  const __bf16* Kbase = Kh + (((size_t)b * NH + hh) * SEQ) * DKH;
  const __bf16* Vbase = Vt + (((size_t)b * NH + hh) * DKH) * SEQ;
  const int*    Mbase = mask + (size_t)b * SEQ * SEQ;

  // Per-thread staging coordinates: 256 16B chunks per tile, 2 per thread.
  int c0 = tid, c1 = tid + 128;
  int kr0 = c0 >> 3, kc0 = (c0 & 7) * 8;
  int kr1 = c1 >> 3, kc1 = (c1 & 7) * 8;
  int vd0 = c0 >> 2, vc0 = (c0 & 3) * 8;
  int vd1 = c1 >> 2, vc1 = (c1 & 3) * 8;

  auto stage = [&](int buf, int s0) {
    async_copy16(Kbase + (size_t)(s0 + kr0) * DKH + kc0, &Ks[buf][kr0 * KS_STRIDE + kc0]);
    async_copy16(Kbase + (size_t)(s0 + kr1) * DKH + kc1, &Ks[buf][kr1 * KS_STRIDE + kc1]);
    async_copy16(Vbase + (size_t)vd0 * SEQ + s0 + vc0,   &Vs[buf][vd0 * VS_STRIDE + vc0]);
    async_copy16(Vbase + (size_t)vd1 * SEQ + s0 + vc1,   &Vs[buf][vd1 * VS_STRIDE + vc1]);
  };

  // Resident Q A-fragments for this lane's row (r = lane&15), K = 0..63
  int qrow = q0 + wave * 16 + nloc;
  const __bf16* Qr = Qbase + (size_t)qrow * DKH;
  FragAB qa0, qa1;
  qa0.h[0] = *(const bf16x8*)(Qr + 16 * half);
  qa0.h[1] = *(const bf16x8*)(Qr + 16 * half + 8);
  qa1.h[0] = *(const bf16x8*)(Qr + 32 + 16 * half);
  qa1.h[1] = *(const bf16x8*)(Qr + 32 + 16 * half + 8);

  f32x8 o[4] = {};
  float mrow[8], lrow[8];
#pragma unroll
  for (int j = 0; j < 8; ++j) { mrow[j] = -3.0e38f; lrow[j] = 0.0f; }
  const float scale = 0.125f;  // 1/sqrt(DKH)

  // Prologue: fill buffer 0.
  stage(0, 0);
  async_wait_and_barrier();

  int cur = 0;
  for (int s0 = 0; s0 < SEQ; s0 += 32) {
    int nxt = cur ^ 1;
    // Kick off next tile's async copies; they land in the other buffer while
    // we compute on the current one.
    if (s0 + 32 < SEQ) stage(nxt, s0 + 32);

    // ---- scores: S[16 q][32 keys] = Q(16x64) @ K^T, two N=16 tiles
    f32x8 sc[2] = {};
#pragma unroll
    for (int nh = 0; nh < 2; ++nh) {
      const __bf16* Kr = &Ks[cur][(16 * nh + nloc) * KS_STRIDE];
      FragAB kb0, kb1;
      kb0.h[0] = *(const bf16x8*)(Kr + 16 * half);
      kb0.h[1] = *(const bf16x8*)(Kr + 16 * half + 8);
      kb1.h[0] = *(const bf16x8*)(Kr + 32 + 16 * half);
      kb1.h[1] = *(const bf16x8*)(Kr + 32 + 16 * half + 8);
      sc[nh] = wmma_bf16(qa0, kb0, sc[nh]);
      sc[nh] = wmma_bf16(qa1, kb1, sc[nh]);
    }

    // ---- scale + mask
#pragma unroll
    for (int nh = 0; nh < 2; ++nh) {
#pragma unroll
      for (int j = 0; j < 8; ++j) {
        int qq = q0 + wave * 16 + j + 8 * half;
        int kk = s0 + 16 * nh + nloc;
        int mv = Mbase[(size_t)qq * SEQ + kk];
        sc[nh][j] = (mv == 0) ? -1.0e9f : sc[nh][j] * scale;
      }
    }

    // ---- online softmax (row reductions across the 16-lane C-layout groups)
    float newm[8], alpha[8];
#pragma unroll
    for (int j = 0; j < 8; ++j) {
      float mx = fmaxf(sc[0][j], sc[1][j]);
#pragma unroll
      for (int d = 1; d < 16; d <<= 1) mx = fmaxf(mx, __shfl_xor(mx, d, 32));
      newm[j]  = fmaxf(mrow[j], mx);
      alpha[j] = __expf(mrow[j] - newm[j]);
      mrow[j]  = newm[j];
    }
#pragma unroll
    for (int j = 0; j < 8; ++j) {
      sc[0][j] = __expf(sc[0][j] - newm[j]);
      sc[1][j] = __expf(sc[1][j] - newm[j]);
      float sm = sc[0][j] + sc[1][j];
#pragma unroll
      for (int d = 1; d < 16; d <<= 1) sm += __shfl_xor(sm, d, 32);
      lrow[j] = lrow[j] * alpha[j] + sm;
#pragma unroll
      for (int t = 0; t < 4; ++t) o[t][j] *= alpha[j];
    }

    // ---- P (C layout) -> LDS -> A layout
    __bf16* Pw = &Ps[wave * 16 * PS_STRIDE];
#pragma unroll
    for (int nh = 0; nh < 2; ++nh) {
#pragma unroll
      for (int j = 0; j < 8; ++j)
        Pw[(j + 8 * half) * PS_STRIDE + 16 * nh + nloc] = (__bf16)sc[nh][j];
    }
    __syncthreads();  // P visible across the wave's lanes (uniform across waves)

    FragAB pa;
    const __bf16* Pr = &Ps[wave * 16 * PS_STRIDE + nloc * PS_STRIDE];
    pa.h[0] = *(const bf16x8*)(Pr + 16 * half);
    pa.h[1] = *(const bf16x8*)(Pr + 16 * half + 8);

    // ---- O += P(16x32) @ V(32x64): 4 N=16 tiles, B from transposed V tile
#pragma unroll
    for (int t = 0; t < 4; ++t) {
      const __bf16* Vr = &Vs[cur][(16 * t + nloc) * VS_STRIDE];
      FragAB vb;
      vb.h[0] = *(const bf16x8*)(Vr + 16 * half);
      vb.h[1] = *(const bf16x8*)(Vr + 16 * half + 8);
      o[t] = wmma_bf16(pa, vb, o[t]);
    }

    // Wait for next tile's async copies + make them visible; also fences the
    // P slab and current buffers before they are reused next iteration.
    async_wait_and_barrier();
    cur = nxt;
  }

  // ---- epilogue: O /= l, write concat layout [b][s][h*DKH+dk] as bf16
#pragma unroll
  for (int t = 0; t < 4; ++t) {
#pragma unroll
    for (int j = 0; j < 8; ++j) {
      float v  = o[t][j] / lrow[j];
      int   qq = q0 + wave * 16 + j + 8 * half;
      int   dk = 16 * t + nloc;
      Obf[((size_t)b * SEQ + qq) * DIM + hh * DKH + dk] = (__bf16)v;
    }
  }
}

// ---------------------------------------------------------------------------
// Launch: convert -> transpose weights -> 3 projection GEMMs -> attention ->
//         output GEMM (fp32 to d_out). All on `stream`, scratch in d_ws.
// ---------------------------------------------------------------------------
extern "C" void kernel_launch(void* const* d_in, const int* in_sizes, int n_in,
                              void* d_out, int out_size, void* d_ws, size_t ws_size,
                              hipStream_t stream) {
  (void)in_sizes; (void)n_in; (void)out_size; (void)ws_size;
  const float* q    = (const float*)d_in[0];
  const float* k    = (const float*)d_in[1];
  const float* v    = (const float*)d_in[2];
  const int*   mask = (const int*)d_in[3];
  const float* Wq   = (const float*)d_in[4];
  const float* bq   = (const float*)d_in[5];
  const float* Wk   = (const float*)d_in[6];
  const float* bk   = (const float*)d_in[7];
  const float* Wv   = (const float*)d_in[8];
  const float* bv   = (const float*)d_in[9];
  const float* Wo   = (const float*)d_in[10];
  const float* bo   = (const float*)d_in[11];

  const size_t ACT = (size_t)BSZ * SEQ * DIM;  // 4M elems
  const size_t WEL = (size_t)DIM * DIM;        // 1M elems
  char* ws = (char*)d_ws;
  __bf16* qb    = (__bf16*)ws; ws += ACT * 2;
  __bf16* kb2   = (__bf16*)ws; ws += ACT * 2;
  __bf16* vb2   = (__bf16*)ws; ws += ACT * 2;
  __bf16* wqt   = (__bf16*)ws; ws += WEL * 2;
  __bf16* wkt   = (__bf16*)ws; ws += WEL * 2;
  __bf16* wvt   = (__bf16*)ws; ws += WEL * 2;
  __bf16* wot   = (__bf16*)ws; ws += WEL * 2;
  __bf16* Qh    = (__bf16*)ws; ws += ACT * 2;
  __bf16* Kh    = (__bf16*)ws; ws += ACT * 2;
  __bf16* Vt    = (__bf16*)ws; ws += ACT * 2;
  __bf16* attnb = (__bf16*)ws; ws += ACT * 2;

  int nact = (int)ACT;
  cvt_f32_bf16_kernel<<<(nact + 255) / 256, 256, 0, stream>>>(q, qb, nact);
  cvt_f32_bf16_kernel<<<(nact + 255) / 256, 256, 0, stream>>>(k, kb2, nact);
  cvt_f32_bf16_kernel<<<(nact + 255) / 256, 256, 0, stream>>>(v, vb2, nact);

  int nw = (int)WEL;
  transpose_bf16_kernel<<<nw / 256, 256, 0, stream>>>(Wq, wqt);
  transpose_bf16_kernel<<<nw / 256, 256, 0, stream>>>(Wk, wkt);
  transpose_bf16_kernel<<<nw / 256, 256, 0, stream>>>(Wv, wvt);
  transpose_bf16_kernel<<<nw / 256, 256, 0, stream>>>(Wo, wot);

  dim3 gg((BSZ * SEQ) / 64, DIM / 64);  // (64, 16)
  gemm_bf16_kernel<<<gg, 128, 0, stream>>>(qb, wqt, bq, Qh, 0);
  gemm_bf16_kernel<<<gg, 128, 0, stream>>>(kb2, wkt, bk, Kh, 0);
  gemm_bf16_kernel<<<gg, 128, 0, stream>>>(vb2, wvt, bv, Vt, 1);

  dim3 ga(SEQ / 64, NH, BSZ);           // (32, 16, 2)
  attn_kernel<<<ga, 128, 0, stream>>>(Qh, Kh, Vt, mask, attnb);

  gemm_bf16_kernel<<<gg, 128, 0, stream>>>(attnb, wot, bo, d_out, 2);
}
